// FiSHAttention_80590766342804
// MI455X (gfx1250) — matrix operand order repaired and checked by reading, not compiled, not run on hardware
//
#include <hip/hip_runtime.h>

typedef __attribute__((ext_vector_type(16))) _Float16 v16h;
typedef __attribute__((ext_vector_type(8)))  float    v8f;

// ---------------------------------------------------------------------------
// WMMA wrapper: D = A(16x32 f16) * B(32x16 f16) + C(16x16 f32)
// ---------------------------------------------------------------------------
__device__ __forceinline__ v8f wmma_f16(v16h a, v16h b, v8f c) {
  return __builtin_amdgcn_wmma_f32_16x16x32_f16(
      /*neg_a=*/false, a, /*neg_b=*/false, b,
      /*c_mod=*/(short)0, c, /*reuse_a=*/false, /*reuse_b=*/false);
}

// ---------------------------------------------------------------------------
// A-fragment from a per-lane base pointer (hlf*16 already folded in):
// one contiguous 16xf16 group at p[0..15] -> two b128 loads.
// Layout: lane L -> row M=L&15, half=L>>4 selects the
// K-subset {half*16+0..15} (matches packed-B contiguous order).
// ---------------------------------------------------------------------------
__device__ __forceinline__ v16h frag_from(const _Float16* p) {
  v16h a;
#pragma unroll
  for (int i = 0; i < 8; ++i) a[i]     = p[i];
#pragma unroll
  for (int i = 0; i < 8; ++i) a[8 + i] = p[8 + i];
  return a;
}

// Convenience A-fragment loader (used where addresses aren't pipelined)
__device__ __forceinline__ v16h load_a_frag(const _Float16* A, int lda,
                                            int row0, int k0) {
  int lane = threadIdx.x & 31;
  return frag_from(A + (long)(row0 + (lane & 15)) * lda + k0 + (lane >> 4) * 16);
}

// ---------------------------------------------------------------------------
// Packed-B layout: per 32x16 tile, lane L's 16 f16 values are contiguous:
//   Bp[(tile*32 + L)*16 + e]  with  k_local = (L>>4)*16 + e,  col = L&15
// ---------------------------------------------------------------------------
__device__ __forceinline__ v16h load_b_packed(const _Float16* Bp, int tile) {
  int lane = threadIdx.x & 31;
  return *(const v16h*)(Bp + (((long)tile * 32 + lane) << 4));
}

// ---------------------------------------------------------------------------
// Ping-pong (unroll-2) pipelined 32x64 wave GEMM: C = A(MxK) * Bpacked.
// Pure pointer stepping (pa += 32, pb += 512 per k-step) -> no per-iteration
// 64-bit address recomputation; two buffer sets alternate -> no register
// rotation copies. KT must be even (24 here). B tile idx = nt*KT + kt.
// ---------------------------------------------------------------------------
__device__ __forceinline__ void wave_gemm_32x64(const _Float16* __restrict__ A,
                                                int lda,
                                                const _Float16* __restrict__ Bp,
                                                int KT, int row0, int colTile0,
                                                v8f acc[2][4]) {
  int lane = threadIdx.x & 31;
  const _Float16* pa = A + (long)(row0 + (lane & 15)) * lda + (lane >> 4) * 16;
  const _Float16* pb = Bp + (((long)colTile0 * KT * 32 + lane) << 4);
  const long arow = (long)16 * lda;   // second row-fragment offset
  const long bcol = (long)KT * 512;   // elements between column strips

#pragma unroll
  for (int rr = 0; rr < 2; ++rr)
#pragma unroll
    for (int t = 0; t < 4; ++t)
#pragma unroll
      for (int r = 0; r < 8; ++r) acc[rr][t][r] = 0.0f;

  v16h a0 = frag_from(pa);
  v16h a1 = frag_from(pa + arow);
  v16h b0[4];
#pragma unroll
  for (int t = 0; t < 4; ++t) b0[t] = *(const v16h*)(pb + t * bcol);

  for (int kt = 0; kt < KT; kt += 2) {
    // load odd k-step fragments (KT even -> always in range)
    pa += 32; pb += 512;
    v16h c0 = frag_from(pa);
    v16h c1 = frag_from(pa + arow);
    v16h b1[4];
#pragma unroll
    for (int t = 0; t < 4; ++t) b1[t] = *(const v16h*)(pb + t * bcol);

#pragma unroll
    for (int t = 0; t < 4; ++t) {
      acc[0][t] = wmma_f16(a0, b0[t], acc[0][t]);
      acc[1][t] = wmma_f16(a1, b0[t], acc[1][t]);
    }

    if (kt + 2 < KT) {  // load next even k-step (skipped on final pair)
      pa += 32; pb += 512;
      a0 = frag_from(pa);
      a1 = frag_from(pa + arow);
#pragma unroll
      for (int t = 0; t < 4; ++t) b0[t] = *(const v16h*)(pb + t * bcol);
    }

#pragma unroll
    for (int t = 0; t < 4; ++t) {
      acc[0][t] = wmma_f16(c0, b1[t], acc[0][t]);
      acc[1][t] = wmma_f16(c1, b1[t], acc[1][t]);
    }
  }
}

// ---------------------------------------------------------------------------
// x: f32 -> f16 row-major (A operand; fragment-friendly as-is)
// ---------------------------------------------------------------------------
__global__ void k_f32_to_f16(const float* __restrict__ s,
                             _Float16* __restrict__ d, long n) {
  long i = (long)blockIdx.x * blockDim.x + threadIdx.x;
  if (i < n) d[i] = (_Float16)s[i];
}

// ---------------------------------------------------------------------------
// Fused transpose + convert + fragment-pack of torch-style weights.
// w: (OutF, InF) f32 row-major; logical B = w^T (K=InF rows, Ncols=OutF).
// ---------------------------------------------------------------------------
__global__ void k_pack_w(const float* __restrict__ w, _Float16* __restrict__ Bp,
                         int OutF, int InF) {
  long f = (long)blockIdx.x * blockDim.x + threadIdx.x;
  if (f >= (long)OutF * InF) return;
  int  e    = (int)(f & 15);
  int  lane = (int)((f >> 4) & 31);
  long tile = f >> 9;
  int  KT   = InF >> 5;
  int  kt   = (int)(tile % KT);
  int  nt   = (int)(tile / KT);
  int  k    = kt * 32 + (lane >> 4) * 16 + e;   // K index (input feature)
  int  o    = nt * 16 + (lane & 15);            // N index (output feature)
  Bp[f] = (_Float16)w[(long)o * InF + k];
}

// ---------------------------------------------------------------------------
// QK GEMM: xh(8192x768) @ w_qk^T(768x512); q -> head-major A layout (scaled
// 1/8), k -> fragment-packed B layout for the attention S-matmul.
// grid (256, 8), block 32 (wave computes 32 rows x 64 cols)
// ---------------------------------------------------------------------------
__global__ __launch_bounds__(32) void k_qk_gemm(const _Float16* __restrict__ xh,
                                                const _Float16* __restrict__ wqkP,
                                                _Float16* __restrict__ Qh,
                                                _Float16* __restrict__ Kp) {
  int row0 = blockIdx.x * 32;
  int col0 = blockIdx.y * 64;
  v8f acc[2][4];
  wave_gemm_32x64(xh, 768, wqkP, 768 / 32, row0, col0 >> 4, acc);

  int lane = threadIdx.x & 31;
  int n = lane & 15, hlf = lane >> 4;
#pragma unroll
  for (int rr = 0; rr < 2; ++rr)
#pragma unroll
    for (int t = 0; t < 4; ++t)
#pragma unroll
      for (int r = 0; r < 8; ++r) {
        int o = col0 + t * 16 + n;
        long m = row0 + rr * 16 + hlf * 8 + r;
        int b = (int)(m >> 10), nn = (int)(m & 1023);
        float v = acc[rr][t][r];
        if (o < 256) {          // Q slice: (B,4,N,64) A-layout, pre-scaled
          int g = o >> 6, d = o & 63;
          Qh[(((long)(b * 4 + g)) * 1024 + nn) * 64 + d] = (_Float16)(v * 0.125f);
        } else {                // K slice: fragment-packed B (K-dim = d, col = j)
          int o2 = o - 256;
          int g = o2 >> 6, d = o2 & 63;
          int kt = d >> 5, kl = d & 31;
          int chlf = kl >> 4, e = kl & 15;
          int ntile = nn >> 4, nc = nn & 15;
          Kp[(((long)(b * 4 + g)) << 16) +
             ((((ntile * 2 + kt) * 32 + chlf * 16 + nc)) << 4) + e] = (_Float16)v;
        }
      }
}

// ---------------------------------------------------------------------------
// V GEMM: xh(8192x768) @ w_v^T -> fragment-packed B layout per (b, h)
// (K-dim = j token index, col = d). grid (256,12), block 32
// ---------------------------------------------------------------------------
__global__ __launch_bounds__(32) void k_v_gemm(const _Float16* __restrict__ xh,
                                               const _Float16* __restrict__ wvP,
                                               _Float16* __restrict__ Vp) {
  int row0 = blockIdx.x * 32;
  int h = blockIdx.y;
  v8f acc[2][4];
  wave_gemm_32x64(xh, 768, wvP, 768 / 32, row0, (h * 64) >> 4, acc);

  int lane = threadIdx.x & 31;
  int n = lane & 15, hlf = lane >> 4;
#pragma unroll
  for (int rr = 0; rr < 2; ++rr)
#pragma unroll
    for (int t = 0; t < 4; ++t)
#pragma unroll
      for (int r = 0; r < 8; ++r) {
        long m = row0 + rr * 16 + hlf * 8 + r;
        int b = (int)(m >> 10), nn = (int)(m & 1023);   // nn = token j
        int kt = nn >> 5, kl = nn & 31;
        int chlf = kl >> 4, e = kl & 15;
        Vp[(((long)(b * 12 + h)) << 16) +
           ((((kt * 4 + t) * 32 + chlf * 16 + n)) << 4) + e] =
            (_Float16)acc[rr][t][r];
      }
}

// ---------------------------------------------------------------------------
// Flash attention with fused head-mixing. One wave per (b, h, 16-row i-tile).
// grid = 8*12*64, block 32.
// ---------------------------------------------------------------------------
__global__ __launch_bounds__(32) void k_attn(const _Float16* __restrict__ Qh,
                                             const _Float16* __restrict__ Kp,
                                             const _Float16* __restrict__ Vp,
                                             const float* __restrict__ mix_logits,
                                             _Float16* __restrict__ Oc) {
  __shared__ _Float16 Plds[16 * 32];

  int id = blockIdx.x;
  int itile = id & 63;
  int h = (id >> 6) % 12;
  int b = id / (64 * 12);
  int i0 = itile * 16;

  int lane = threadIdx.x & 31;
  int n = lane & 15, hlf = lane >> 4;

  // softmax over the 4 mix logits for this head
  float ml[4], mmax = -1e30f, msum = 0.f;
#pragma unroll
  for (int g = 0; g < 4; ++g) { ml[g] = mix_logits[h * 4 + g]; mmax = fmaxf(mmax, ml[g]); }
#pragma unroll
  for (int g = 0; g < 4; ++g) { ml[g] = __expf(ml[g] - mmax); msum += ml[g]; }
#pragma unroll
  for (int g = 0; g < 4; ++g) ml[g] /= msum;

  // Q fragments, pre-scaled by mix weight (1/8 folded in at QK GEMM)
  v16h aq[4][2];
#pragma unroll
  for (int g = 0; g < 4; ++g) {
    const _Float16* Qg = Qh + (((long)(b * 4 + g)) * 1024 + i0) * 64;
    _Float16 s = (_Float16)ml[g];
#pragma unroll
    for (int c = 0; c < 2; ++c) {
      v16h f = load_a_frag(Qg, 64, 0, c * 32);
#pragma unroll
      for (int e = 0; e < 16; ++e) f[e] = f[e] * s;
      aq[g][c] = f;
    }
  }

  float mrow[8], lrow[8];
  v8f ov[4];
#pragma unroll
  for (int r = 0; r < 8; ++r) { mrow[r] = -1e30f; lrow[r] = 0.f; }
#pragma unroll
  for (int t = 0; t < 4; ++t)
#pragma unroll
    for (int r = 0; r < 8; ++r) ov[t][r] = 0.f;

  // per-lane stepped fragment pointers (1 KB per 32x16 tile)
  const _Float16* pk[4];
#pragma unroll
  for (int g = 0; g < 4; ++g)
    pk[g] = Kp + (((long)(b * 4 + g)) << 16) + ((long)lane << 4);
  const _Float16* pv = Vp + (((long)(b * 12 + h)) << 16) + ((long)lane << 4);

  for (int j0 = 0; j0 < 1024; j0 += 32) {
    if (j0 + 32 < 1024) {  // prefetch next K tile group of head 0's stream
      __builtin_prefetch(pk[0] + 4 * 512, 0, 0);
    }
    // mixed logits: S = sum_g mix[h,g] * q_g k_g^T  (two 16x16 tiles)
    v8f s[2];
#pragma unroll
    for (int jj = 0; jj < 2; ++jj) {
#pragma unroll
      for (int r = 0; r < 8; ++r) s[jj][r] = 0.f;
#pragma unroll
      for (int g = 0; g < 4; ++g)
#pragma unroll
        for (int c = 0; c < 2; ++c) {
          v16h bf = *(const v16h*)(pk[g] + (jj * 2 + c) * 512);
          s[jj] = wmma_f16(aq[g][c], bf, s[jj]);
        }
    }
#pragma unroll
    for (int g = 0; g < 4; ++g) pk[g] += 4 * 512;

    // Hoist V B-fragments: issue loads now so their latency hides under the
    // softmax VALU/shfl section below (first consumed after it).
    v16h bv[4];
#pragma unroll
    for (int t = 0; t < 4; ++t) bv[t] = *(const v16h*)(pv + t * 512);
    pv += 4 * 512;

    // online softmax (rows = i, butterfly over 16 lanes of a half)
    float mnew[8], alpha[8];
#pragma unroll
    for (int r = 0; r < 8; ++r) {
      float v = fmaxf(s[0][r], s[1][r]);
#pragma unroll
      for (int off = 8; off >= 1; off >>= 1) v = fmaxf(v, __shfl_xor(v, off, 32));
      mnew[r]  = fmaxf(mrow[r], v);
      alpha[r] = __expf(mrow[r] - mnew[r]);
    }
#pragma unroll
    for (int r = 0; r < 8; ++r) {
      float p0 = __expf(s[0][r] - mnew[r]);
      float p1 = __expf(s[1][r] - mnew[r]);
      Plds[(hlf * 8 + r) * 32 + n]      = (_Float16)p0;
      Plds[(hlf * 8 + r) * 32 + 16 + n] = (_Float16)p1;
      float rs = p0 + p1;
#pragma unroll
      for (int off = 8; off >= 1; off >>= 1) rs += __shfl_xor(rs, off, 32);
      lrow[r] = lrow[r] * alpha[r] + rs;
      mrow[r] = mnew[r];
    }
    __syncthreads();  // C-layout -> A-layout re-distribution via LDS

    v16h pa = load_a_frag(Plds, 32, 0, 0);
#pragma unroll
    for (int t = 0; t < 4; ++t) {
#pragma unroll
      for (int r = 0; r < 8; ++r) ov[t][r] *= alpha[r];
      ov[t] = wmma_f16(pa, bv[t], ov[t]);
    }
    __syncthreads();  // protect Plds before next iteration's stores
  }

  long row = (long)b * 1024 + i0 + hlf * 8;
#pragma unroll
  for (int t = 0; t < 4; ++t)
#pragma unroll
    for (int r = 0; r < 8; ++r) {
      float val = ov[t][r] / lrow[r];
      Oc[(row + r) * 768 + h * 64 + t * 16 + n] = (_Float16)val;
    }
}

// ---------------------------------------------------------------------------
// Proj GEMM: Oc(8192x768) @ w_proj^T + bias -> f32 out. grid (256,12)
// ---------------------------------------------------------------------------
__global__ __launch_bounds__(32) void k_proj_gemm(const _Float16* __restrict__ Oc,
                                                  const _Float16* __restrict__ wpP,
                                                  const float* __restrict__ bias,
                                                  float* __restrict__ out) {
  int row0 = blockIdx.x * 32;
  int col0 = blockIdx.y * 64;
  v8f acc[2][4];
  wave_gemm_32x64(Oc, 768, wpP, 768 / 32, row0, col0 >> 4, acc);

  int lane = threadIdx.x & 31;
  int n = lane & 15, hlf = lane >> 4;
#pragma unroll
  for (int rr = 0; rr < 2; ++rr)
#pragma unroll
    for (int t = 0; t < 4; ++t)
#pragma unroll
      for (int r = 0; r < 8; ++r) {
        int col = col0 + t * 16 + n;
        out[(long)(row0 + rr * 16 + hlf * 8 + r) * 768 + col] =
            acc[rr][t][r] + bias[col];
      }
}

// ---------------------------------------------------------------------------
extern "C" void kernel_launch(void* const* d_in, const int* in_sizes, int n_in,
                              void* d_out, int out_size, void* d_ws, size_t ws_size,
                              hipStream_t stream) {
  (void)in_sizes; (void)n_in; (void)out_size; (void)ws_size;

  const float* x          = (const float*)d_in[0];  // (8,1024,768)
  const float* w_qkv      = (const float*)d_in[1];  // (768,768) - rows [0,512) used
  const float* mix_logits = (const float*)d_in[2];  // (12,4)
  const float* w_v        = (const float*)d_in[3];  // (768,768)
  const float* w_proj     = (const float*)d_in[4];  // (768,768)
  const float* b_proj     = (const float*)d_in[5];  // (768,)
  float* out              = (float*)d_out;          // (8,1024,768)

  const long M = 8192;  // B*N

  char* ws = (char*)d_ws;
  _Float16* xh   = (_Float16*)ws; ws += M * 768 * 2;              // f16 x (A layout)
  _Float16* wqkP = (_Float16*)ws; ws += 768l * 512 * 2;           // packed B (768x512)
  _Float16* wvP  = (_Float16*)ws; ws += 768l * 768 * 2;           // packed B
  _Float16* wpP  = (_Float16*)ws; ws += 768l * 768 * 2;           // packed B
  _Float16* Qh   = (_Float16*)ws; ws += 8l * 4 * 1024 * 64 * 2;   // (B,4,N,64) A layout
  _Float16* Kp   = (_Float16*)ws; ws += 8l * 4 * 64 * 1024 * 2;   // packed B per (b,g)
  _Float16* Vp   = (_Float16*)ws; ws += 8l * 12 * 1024 * 64 * 2;  // packed B per (b,h)
  _Float16* Oc   = (_Float16*)ws; ws += M * 768 * 2;              // attention out (A)

  {
    long n = M * 768;
    k_f32_to_f16<<<dim3((unsigned)((n + 255) / 256)), 256, 0, stream>>>(x, xh, n);
  }
  k_pack_w<<<dim3((512 * 768 + 255) / 256), 256, 0, stream>>>(w_qkv, wqkP, 512, 768);
  k_pack_w<<<dim3((768 * 768 + 255) / 256), 256, 0, stream>>>(w_v,    wvP,  768, 768);
  k_pack_w<<<dim3((768 * 768 + 255) / 256), 256, 0, stream>>>(w_proj, wpP,  768, 768);

  // QK projection (q,k slices only — v-slice of w_qkv is dead in the reference)
  k_qk_gemm<<<dim3(256, 8), 32, 0, stream>>>(xh, wqkP, Qh, Kp);
  // V projection
  k_v_gemm<<<dim3(256, 12), 32, 0, stream>>>(xh, wvP, Vp);
  // fused mixed-head flash attention
  k_attn<<<dim3(8 * 12 * 64), 32, 0, stream>>>(Qh, Kp, Vp, mix_logits, Oc);
  // output projection + bias
  k_proj_gemm<<<dim3(256, 12), 32, 0, stream>>>(Oc, wpP, b_proj, out);
}
